// MultiHeadAttention_75754633167392
// MI455X (gfx1250) — matrix operand order repaired and checked by reading, not compiled, not run on hardware
//
#include <hip/hip_runtime.h>
#include <hip/hip_bf16.h>

// MI455X (gfx1250) implementation.
// ~86 GFLOP total, 319 MB mandatory output stores (~14us @ 23.3 TB/s).
// All big matmuls use v_wmma_f32_16x16x32_bf16 (fp32 accumulate).
// Workspace use: ~40.5 MB.

constexpr int B_  = 8;
constexpr int L_  = 1024;
constexpr int D_  = 512;
constexpr int H_  = 8;
constexpr int DK_ = 64;
constexpr int KW_ = 7;
constexpr int PAD_ = 3;

typedef __attribute__((ext_vector_type(16))) __bf16 v16bf;
typedef __attribute__((ext_vector_type(8)))  float  v8f;
typedef __attribute__((ext_vector_type(2)))  float  f32x2;
typedef __attribute__((ext_vector_type(2)))  __bf16 bf16x2;

union BF16x16 {
  v16bf bf;
  bf16x2 h2[8];
  unsigned short u16[16];
  unsigned int u32[8];
};

// packed f32->bf16 conversion: 8 floats -> elements [2*base .. 2*base+7]
__device__ __forceinline__ void cvt8(BF16x16& dst, int base, const float* p) {
#pragma unroll
  for (int e = 0; e < 4; ++e) {
    f32x2 t;
    t[0] = p[2 * e];
    t[1] = p[2 * e + 1];
    dst.h2[base + e] = __builtin_convertvector(t, bf16x2);
  }
}

__device__ __forceinline__ unsigned short f2bf(float f) {
  union { __bf16 b; unsigned short u; } cv;
  cv.b = (__bf16)f;
  return cv.u;
}
__device__ __forceinline__ float bf2f(unsigned short u) {
  return __uint_as_float(((unsigned int)u) << 16);
}
__device__ __forceinline__ v8f wmma_bf16(v16bf a, v16bf b, v8f c) {
  return __builtin_amdgcn_wmma_f32_16x16x32_bf16(false, a, false, b, (short)0, c,
                                                 false, false);
}

// ---------------------------------------------------------------------------
// Kernel 1: q_s/k_s/v_s projections.  out = X @ W^T + b, stored bf16.
//   q_s, k_s : [B,H,L,DK] row-major      (A-operand of later GEMMs)
//   v_t      : [B,H,DK,L] (transposed)   (B-operand of attn@V, contiguous)
// grid (M/16, 1, 3), block 256. Each wave owns 4 N-tiles: A-fragment is
// converted once per k-step and reused by 4 WMMAs.
// ---------------------------------------------------------------------------
__global__ __launch_bounds__(256) void qkv_proj_kernel(
    const float* __restrict__ q, const float* __restrict__ k,
    const float* __restrict__ v,
    const float* __restrict__ Wq, const float* __restrict__ bq,
    const float* __restrict__ Wk, const float* __restrict__ bk,
    const float* __restrict__ Wv, const float* __restrict__ bv,
    unsigned short* __restrict__ qs, unsigned short* __restrict__ ks,
    unsigned short* __restrict__ vt)
{
  const int lane = threadIdx.x & 31;
  const int wave = threadIdx.x >> 5;
  const int hi   = lane >> 4;
  const int lan  = lane & 15;
  const int m0 = blockIdx.x * 16;
  const int which = blockIdx.z;

  const float* X    = which == 0 ? q  : (which == 1 ? k  : v);
  const float* W    = which == 0 ? Wq : (which == 1 ? Wk : Wv);
  const float* bias = which == 0 ? bq : (which == 1 ? bk : bv);

  const size_t rowA = (size_t)(m0 + lan) * D_;

  v8f zero = {};
  v8f acc[4];
#pragma unroll
  for (int u = 0; u < 4; ++u) acc[u] = zero;

  for (int k0 = 0; k0 < D_; k0 += 32) {
    BF16x16 a;
    const float* pa = X + rowA + k0 + hi * 16;
    cvt8(a, 0, pa);
    cvt8(a, 4, pa + 8);
#pragma unroll
    for (int u = 0; u < 4; ++u) {
      const int n0 = wave * 64 + u * 16;
      BF16x16 bm;
      const float* pb = W + (size_t)(n0 + lan) * D_ + k0 + hi * 16;
      cvt8(bm, 0, pb);
      cvt8(bm, 4, pb + 8);
      acc[u] = wmma_bf16(a.bf, bm.bf, acc[u]);
    }
  }

#pragma unroll
  for (int u = 0; u < 4; ++u) {
    const int n = wave * 64 + u * 16 + lan;
    const int h = n / DK_, d = n % DK_;
    const float bn = bias[n];
#pragma unroll
    for (int r = 0; r < 8; ++r) {
      const int m = m0 + r + hi * 8;
      const int b = m >> 10, i = m & (L_ - 1);
      const unsigned short val = f2bf(acc[u][r] + bn);
      if (which == 2)
        vt[(((size_t)b * H_ + h) * DK_ + d) * L_ + i] = val;
      else if (which == 0)
        qs[(((size_t)b * H_ + h) * L_ + i) * DK_ + d] = val;
      else
        ks[(((size_t)b * H_ + h) * L_ + i) * DK_ + d] = val;
    }
  }
}

// ---------------------------------------------------------------------------
// Kernel 2: scalar row bias  bias_q[b,h,i] = q_s . Wqb[h] + bqb[h]
//           scalar col bias  biasK[b,h,j]  = sum_t,c bker[h,c,t]*k_pad[j+t,c]
// ---------------------------------------------------------------------------
__global__ __launch_bounds__(256) void bias_kernel(
    const unsigned short* __restrict__ qs, const unsigned short* __restrict__ ks,
    const float* __restrict__ Wqb, const float* __restrict__ bqb,
    const float* __restrict__ bker,
    float* __restrict__ bias_q, float* __restrict__ biasK)
{
  const int idx = blockIdx.x * blockDim.x + threadIdx.x;   // B*H*L
  if (idx >= B_ * H_ * L_) return;
  const int bh = idx / L_;
  const int h  = bh & (H_ - 1);
  const int i  = idx % L_;

  const unsigned short* qrow = qs + (size_t)idx * DK_;
  float s = 0.0f;
#pragma unroll
  for (int d = 0; d < DK_; ++d) s += bf2f(qrow[d]) * Wqb[h * DK_ + d];
  bias_q[idx] = s + bqb[h];

  float sk = 0.0f;
  for (int t = 0; t < KW_; ++t) {
    const int j = i + t - PAD_;
    if (j >= 0 && j < L_) {
      const unsigned short* krow = ks + ((size_t)bh * L_ + j) * DK_;
#pragma unroll
      for (int c = 0; c < DK_; ++c)
        sk += bker[(h * DK_ + c) * KW_ + t] * bf2f(krow[c]);
    }
  }
  biasK[idx] = sk;
}

// ---------------------------------------------------------------------------
// Kernel 3: fused attention.
// One workgroup = (b, h, 16-row i-tile). 8 waves. LDS: 14KB Qext + 64KB
// logits + 4KB ctx (well under 320KB/WGP).
//   Phase 0: Qext(16x448) = q_s(16x64) @ Wker_h^T  (WMMA) -> LDS bf16
//   Phase 1: logits(16x1024) = Qext @ Kshift^T + biases    (WMMA, K=448,
//            K-outer loop, 8 live accumulators reuse the A fragment)
//   Phase 2: row softmax (wave32 shfl), write attn / one_head
//   Phase 3: ctx(16x64) = attn @ v                          (WMMA, ds_add_f32)
// ---------------------------------------------------------------------------
__global__ __launch_bounds__(256) void attn_kernel(
    const unsigned short* __restrict__ qs, const unsigned short* __restrict__ ks,
    const unsigned short* __restrict__ vt,
    const float* __restrict__ Wker,
    const float* __restrict__ bias_q, const float* __restrict__ biasK,
    const float* __restrict__ bias_b,
    const unsigned char* __restrict__ mask,
    float* __restrict__ attn_out, float* __restrict__ one_head_out,
    float* __restrict__ ctx_out)
{
  __shared__ alignas(16) unsigned short sA[16 * 448];   // Qext tile (bf16)
  __shared__ alignas(16) float sLog[16 * 1024];         // logits / probs
  __shared__ alignas(16) float sCtx[16 * 64];           // ctx accumulator

  const int tid  = threadIdx.x;
  const int lane = tid & 31;
  const int wave = tid >> 5;
  const int hi   = lane >> 4;
  const int lan  = lane & 15;
  const int i0 = blockIdx.x * 16;
  const int bh = blockIdx.y;
  const int h  = bh & (H_ - 1);
  const int b  = bh >> 3;

  v8f zero = {};

  for (int idx = tid; idx < 16 * 64; idx += 256) sCtx[idx] = 0.0f;

  // ---- Phase 0: Qext tile (A fragments hoisted; nt-invariant) ----
  {
    const unsigned short* qrow = qs + ((size_t)bh * L_ + i0 + lan) * DK_;
    BF16x16 qa0, qa1;
    const unsigned int* p0 = (const unsigned int*)(qrow + hi * 16);
    const unsigned int* p1 = (const unsigned int*)(qrow + 32 + hi * 16);
#pragma unroll
    for (int e = 0; e < 4; ++e) { qa0.u32[e] = p0[e]; qa0.u32[4 + e] = p0[4 + e]; }
#pragma unroll
    for (int e = 0; e < 4; ++e) { qa1.u32[e] = p1[e]; qa1.u32[4 + e] = p1[4 + e]; }

    for (int nt = wave; nt < 28; nt += 8) {
      const int n0 = nt * 16;
      const int t  = n0 / DK_;            // a 16-wide tile stays within one t
      const int c  = (n0 % DK_) + lan;
      const float* pb = Wker + (((size_t)h * DK_ + c) * KW_ + t) * DK_ + hi * 16;
      BF16x16 bm;
      v8f acc = zero;
      cvt8(bm, 0, pb);
      cvt8(bm, 4, pb + 8);
      acc = wmma_bf16(qa0.bf, bm.bf, acc);
      cvt8(bm, 0, pb + 32);
      cvt8(bm, 4, pb + 40);
      acc = wmma_bf16(qa1.bf, bm.bf, acc);
#pragma unroll
      for (int r = 0; r < 8; ++r)
        sA[(r + hi * 8) * 448 + n0 + lan] = f2bf(acc[r]);
    }
  }
  __syncthreads();

  float bq8[8];
#pragma unroll
  for (int r = 0; r < 8; ++r)
    bq8[r] = bias_q[(size_t)bh * L_ + i0 + r + hi * 8] + bias_b[h];

  // ---- Phase 1: logits, 128 columns per wave, K-outer for A reuse ----
  {
    v8f acc[8];
#pragma unroll
    for (int jt = 0; jt < 8; ++jt) acc[jt] = zero;

    for (int kk = 0; kk < 14; ++kk) {
      const int k0 = kk * 32;
      const int t  = kk >> 1;
      const int cb = (kk & 1) * 32;
      BF16x16 a;
      const unsigned int* pa =
          (const unsigned int*)(sA + lan * 448 + k0 + hi * 16);   // LDS reads
#pragma unroll
      for (int e = 0; e < 4; ++e) { a.u32[e] = pa[e]; a.u32[4 + e] = pa[4 + e]; }

#pragma unroll
      for (int jt = 0; jt < 8; ++jt) {
        const int jn = wave * 128 + jt * 16 + lan + t - PAD_;    // shifted K
        BF16x16 bm;
        if (jn >= 0 && jn < L_) {
          const unsigned int* pb = (const unsigned int*)(
              ks + ((size_t)bh * L_ + jn) * DK_ + cb + hi * 16);
#pragma unroll
          for (int e = 0; e < 8; ++e) bm.u32[e] = pb[e];
        } else {
#pragma unroll
          for (int e = 0; e < 8; ++e) bm.u32[e] = 0u;
        }
        acc[jt] = wmma_bf16(a.bf, bm.bf, acc[jt]);
      }
    }

#pragma unroll
    for (int jt = 0; jt < 8; ++jt) {
      const int j = wave * 128 + jt * 16 + lan;
      const float bkj = biasK[(size_t)bh * L_ + j];
      const unsigned char* mcol = mask + ((size_t)b * L_ + i0) * L_ + j;
#pragma unroll
      for (int r = 0; r < 8; ++r) {
        const int ml = r + hi * 8;
        float val = (acc[jt][r] + bq8[r] + bkj) * 0.125f;   // 1/sqrt(DK)
        if (mcol[(size_t)ml * L_]) val = -__builtin_inff();
        sLog[ml * 1024 + j] = val;
      }
    }
  }
  __syncthreads();

  // ---- Phase 2: softmax, one wave per 2 rows ----
  for (int rr = 0; rr < 2; ++rr) {
    const int row = wave * 2 + rr;
    float* rowp = sLog + row * 1024;
    float mx = -__builtin_inff();
    for (int cc = lane; cc < 1024; cc += 32) mx = fmaxf(mx, rowp[cc]);
#pragma unroll
    for (int off = 16; off > 0; off >>= 1)
      mx = fmaxf(mx, __shfl_xor(mx, off, 32));
    float s = 0.0f;
    for (int cc = lane; cc < 1024; cc += 32) {
      const float e = __expf(rowp[cc] - mx);
      rowp[cc] = e;
      s += e;
    }
#pragma unroll
    for (int off = 16; off > 0; off >>= 1) s += __shfl_xor(s, off, 32);
    const float inv = 1.0f / s;
    const int i = i0 + row;
    float* ap = attn_out + ((size_t)bh * L_ + i) * L_;
    float* op = (h == 0) ? (one_head_out + ((size_t)b * L_ + i) * L_) : nullptr;
    for (int cc = lane; cc < 1024; cc += 32) {
      const float p = rowp[cc] * inv;
      rowp[cc] = p;
      ap[cc] = p;
      if (op) op[cc] = p;
    }
  }
  __syncthreads();

  // ---- Phase 3: ctx = attn @ v, K-outer for A reuse, 4 accumulators ----
  {
    v8f cacc[4];
#pragma unroll
    for (int nt = 0; nt < 4; ++nt) cacc[nt] = zero;

#pragma unroll
    for (int ki = 0; ki < 4; ++ki) {
      const int k0 = wave * 128 + ki * 32;
      BF16x16 a;
      const float* pa = sLog + lan * 1024 + k0 + hi * 16;
      cvt8(a, 0, pa);
      cvt8(a, 4, pa + 8);
#pragma unroll
      for (int nt = 0; nt < 4; ++nt) {
        BF16x16 bm;
        const unsigned int* pb = (const unsigned int*)(
            vt + ((size_t)bh * DK_ + nt * 16 + lan) * L_ + k0 + hi * 16);
#pragma unroll
        for (int e = 0; e < 8; ++e) bm.u32[e] = pb[e];
        cacc[nt] = wmma_bf16(a.bf, bm.bf, cacc[nt]);
      }
    }
#pragma unroll
    for (int nt = 0; nt < 4; ++nt)
#pragma unroll
      for (int r = 0; r < 8; ++r)
        atomicAdd(&sCtx[(r + hi * 8) * 64 + nt * 16 + lan], cacc[nt][r]);
  }
  __syncthreads();

  // ---- Phase 4: write ctx merged to [B,L,D] ----
  for (int idx = tid; idx < 16 * 64; idx += 256) {
    const int r = idx >> 6, d = idx & 63;
    ctx_out[((size_t)b * L_ + i0 + r) * D_ + h * DK_ + d] = sCtx[idx];
  }
}

// ---------------------------------------------------------------------------
// Kernel 4: output = ctx @ Wproj^T + bproj  (fp32 out)
// grid (M/16), block 256. Same 4-N-tile A-reuse layout as kernel 1.
// ---------------------------------------------------------------------------
__global__ __launch_bounds__(256) void out_proj_kernel(
    const float* __restrict__ ctx, const float* __restrict__ Wproj,
    const float* __restrict__ bproj, float* __restrict__ out)
{
  const int lane = threadIdx.x & 31;
  const int wave = threadIdx.x >> 5;
  const int hi   = lane >> 4;
  const int lan  = lane & 15;
  const int m0 = blockIdx.x * 16;
  const size_t rowA = (size_t)(m0 + lan) * D_;

  v8f zero = {};
  v8f acc[4];
#pragma unroll
  for (int u = 0; u < 4; ++u) acc[u] = zero;

  for (int k0 = 0; k0 < D_; k0 += 32) {
    BF16x16 a;
    const float* pa = ctx + rowA + k0 + hi * 16;
    cvt8(a, 0, pa);
    cvt8(a, 4, pa + 8);
#pragma unroll
    for (int u = 0; u < 4; ++u) {
      const int n0 = wave * 64 + u * 16;
      BF16x16 bm;
      const float* pb = Wproj + (size_t)(n0 + lan) * D_ + k0 + hi * 16;
      cvt8(bm, 0, pb);
      cvt8(bm, 4, pb + 8);
      acc[u] = wmma_bf16(a.bf, bm.bf, acc[u]);
    }
  }
#pragma unroll
  for (int u = 0; u < 4; ++u) {
    const int n = wave * 64 + u * 16 + lan;
    const float bn = bproj[n];
#pragma unroll
    for (int r = 0; r < 8; ++r)
      out[(size_t)(m0 + r + hi * 8) * D_ + n] = acc[u][r] + bn;
  }
}

// ---------------------------------------------------------------------------
extern "C" void kernel_launch(void* const* d_in, const int* in_sizes, int n_in,
                              void* d_out, int out_size, void* d_ws,
                              size_t ws_size, hipStream_t stream)
{
  const float* q  = (const float*)d_in[0];
  const float* k  = (const float*)d_in[1];
  const float* v  = (const float*)d_in[2];
  const unsigned char* mask = (const unsigned char*)d_in[3];
  const float* Wq = (const float*)d_in[4];
  const float* bq = (const float*)d_in[5];
  const float* Wk = (const float*)d_in[6];
  const float* bk = (const float*)d_in[7];
  const float* Wv = (const float*)d_in[8];
  const float* bv = (const float*)d_in[9];
  const float* Wker   = (const float*)d_in[10];
  const float* bker   = (const float*)d_in[11];
  const float* Wqb    = (const float*)d_in[12];
  const float* bqb    = (const float*)d_in[13];
  const float* bias_b = (const float*)d_in[14];
  const float* Wproj  = (const float*)d_in[15];
  const float* bproj  = (const float*)d_in[16];

  float* out      = (float*)d_out;                    // [B,L,D]
  float* attn     = out  + (size_t)B_ * L_ * D_;      // [B,H,L,L]
  float* one_head = attn + (size_t)B_ * H_ * L_ * L_; // [B,L,L]

  char* ws = (char*)d_ws;
  unsigned short* qs  = (unsigned short*)(ws);             //  8 MiB [B,H,L,DK] bf16
  unsigned short* ksb = (unsigned short*)(ws +  8388608);  //  8 MiB [B,H,L,DK] bf16
  unsigned short* vt  = (unsigned short*)(ws + 16777216);  //  8 MiB [B,H,DK,L] bf16
  float* bias_q = (float*)(ws + 25165824);                 // 256 KiB
  float* biasK  = (float*)(ws + 25427968);                 // 256 KiB
  float* ctx    = (float*)(ws + 25690112);                 // 16 MiB [B,L,D] f32
  // total workspace: ~40.5 MiB

  qkv_proj_kernel<<<dim3(B_ * L_ / 16, 1, 3), 256, 0, stream>>>(
      q, k, v, Wq, bq, Wk, bk, Wv, bv, qs, ksb, vt);
  bias_kernel<<<dim3(B_ * H_ * L_ / 256), 256, 0, stream>>>(
      qs, ksb, Wqb, bqb, bker, bias_q, biasK);
  attn_kernel<<<dim3(L_ / 16, B_ * H_), 256, 0, stream>>>(
      qs, ksb, vt, Wker, bias_q, biasK, bias_b, mask, attn, one_head, ctx);
  out_proj_kernel<<<dim3(B_ * L_ / 16), 256, 0, stream>>>(
      ctx, Wproj, bproj, out);
}
